// PatchCore_12163347382640
// MI455X (gfx1250) — compile-verified, matrix operand order checked by reading, not run on hardware
//
#include <hip/hip_runtime.h>
#include <hip/hip_bf16.h>
#include <math.h>

// ---------------- problem constants ----------------
#define B_IMG 8
#define FH_ 28
#define FW_ 28
#define IMG_ 224
#define C_DIM 1024
#define N_FEAT (B_IMG*FH_*FW_)     // 6272 = 49*128
#define M_MEM 50000
#define M_PAD 50048                // 391*128
#define KS_ 17
#define PAD_R 8
#define SIGMA_ 4.0f

// ---------------- GEMM tiling ----------------
#define BN 128
#define BM 128
#define BK 32
#define LDA 40   // LDS row stride (elements), padded to spread banks, keeps 16B align

typedef __attribute__((ext_vector_type(16))) __bf16 v16bf;
typedef __attribute__((ext_vector_type(8)))  float  v8f;

union Frag { uint4 u[2]; v16bf v; };

// CDNA5 async copy: global -> LDS, 16B per lane, no data VGPRs, tracked by ASYNCcnt.
// vDST = 32-bit LDS byte offset (low 32 bits of a generic __shared__ pointer),
// vADDR = 64-bit global address (VGPR pair), SADDR = off  (GV mode, ISA 15.18.3 op 98).
__device__ __forceinline__ void async_ld16(unsigned lds_off, const void* g) {
  asm volatile("global_load_async_to_lds_b128 %0, %1, off"
               :: "v"(lds_off), "v"((unsigned long long)(uintptr_t)g)
               : "memory");
}
__device__ __forceinline__ void wait_async0() {
  asm volatile("s_wait_asynccnt 0x0" ::: "memory");
}

// ---------------- prep: fp32 -> bf16 hi/lo (lo optional) ----------------
__global__ void k_convert(const float* __restrict__ src,
                          __hip_bfloat16* __restrict__ hi,
                          __hip_bfloat16* __restrict__ lo,
                          long src_elems, long pad_elems) {
  long i4 = (long)blockIdx.x * blockDim.x + threadIdx.x;
  long e = i4 * 4;
  if (e >= pad_elems) return;
  float4 v = make_float4(0.f, 0.f, 0.f, 0.f);
  if (e < src_elems) v = ((const float4*)src)[i4];
  float x[4] = {v.x, v.y, v.z, v.w};
  __hip_bfloat16 h[4], l[4];
#pragma unroll
  for (int j = 0; j < 4; j++) {
    h[j] = __float2bfloat16(x[j]);
    float hf = __bfloat162float(h[j]);
    l[j] = __float2bfloat16(x[j] - hf);
  }
  *(uint2*)&hi[e] = *(uint2*)h;
  if (lo) *(uint2*)&lo[e] = *(uint2*)l;
}

// ---------------- prep: row squared norms ----------------
__global__ void k_rownorm(const float* __restrict__ src, float* __restrict__ out,
                          int rows_src, float padval) {
  int row = blockIdx.x;
  float s = 0.f;
  if (row < rows_src) {
    const float* p = src + (size_t)row * C_DIM;
    for (int k = threadIdx.x; k < C_DIM; k += blockDim.x) { float v = p[k]; s += v * v; }
  }
#pragma unroll
  for (int off = 16; off >= 1; off >>= 1) s += __shfl_xor(s, off, 32);
  __shared__ float ls[4];
  int w = threadIdx.x >> 5;
  if ((threadIdx.x & 31) == 0) ls[w] = s;
  __syncthreads();
  if (threadIdx.x == 0) {
    float t = ls[0] + ls[1] + ls[2] + ls[3];
    out[row] = (row < rows_src) ? t : padval;
  }
}

__global__ void k_init_min(unsigned* __restrict__ m, int n) {
  int i = blockIdx.x * blockDim.x + threadIdx.x;
  if (i < n) m[i] = 0x7F7FFFFFu;  // FLT_MAX bits (nonneg floats order as uints)
}

// ---------------- fused GEMM + min over M (bf16x3 split, templated) ----------------
// D-tile layout assumption (ISA 7.12.2): C/D VGPR r, lane l -> row M=r+8*(l>>4), col N=l&15.
// A frag (16x32 bf16): lane half 0 holds K {0..7,16..23}, half 1 holds K {8..15,24..31}.
// B frag assumed mirrored (lane = column, same K split).
// LDS double buffer; next K-slab staged via global_load_async_to_lds_b128 (no data VGPRs,
// no spills), WMMAs run on the current buffer meanwhile; s_wait_asynccnt 0 before the
// single per-iteration barrier publishes the freshly written buffer.
template<int SPLIT>
__global__ __launch_bounds__(256, 2) void k_gemm_min(
    const __hip_bfloat16* __restrict__ fh, const __hip_bfloat16* __restrict__ fl,
    const __hip_bfloat16* __restrict__ mh, const __hip_bfloat16* __restrict__ ml,
    const float* __restrict__ f2, const float* __restrict__ m2,
    unsigned* __restrict__ minbits)
{
  __shared__ __align__(16) unsigned short sAh[2][BN * LDA], sAl[2][BN * LDA];
  __shared__ __align__(16) unsigned short sBh[2][BM * LDA], sBl[2][BM * LDA];

  const int t = threadIdx.x;
  const int nBase = blockIdx.x * BN;
  const int mBase = blockIdx.y * BM;
  const int l  = t & 31, w = t >> 5;
  const int l2 = l & 15, hf = l >> 4;
  const int wn = w & 3,  wm = w >> 2;   // 4x2 wave grid: 2 n-tiles x 4 m-tiles each

  // per-thread staging chunk coordinates (2 x 16B chunks per buffer)
  size_t baseA[2], baseB[2];
  int soff[2];
#pragma unroll
  for (int cc = 0; cc < 2; cc++) {
    int c = t + cc * 256;
    int row = c >> 2;
    int ko  = (c & 3) * 8;
    baseA[cc] = (size_t)(nBase + row) * C_DIM + ko;
    baseB[cc] = (size_t)(mBase + row) * C_DIM + ko;
    soff[cc]  = row * LDA + ko;
  }

  auto stage = [&](int buf, int k0) {
#pragma unroll
    for (int cc = 0; cc < 2; cc++) {
      async_ld16((unsigned)(uintptr_t)&sAh[buf][soff[cc]], &fh[baseA[cc] + k0]);
      async_ld16((unsigned)(uintptr_t)&sBh[buf][soff[cc]], &mh[baseB[cc] + k0]);
      if (SPLIT) {
        async_ld16((unsigned)(uintptr_t)&sAl[buf][soff[cc]], &fl[baseA[cc] + k0]);
        async_ld16((unsigned)(uintptr_t)&sBl[buf][soff[cc]], &ml[baseB[cc] + k0]);
      }
    }
  };

  const v8f vzero = {0.f,0.f,0.f,0.f,0.f,0.f,0.f,0.f};
  v8f acc[2][4];
#pragma unroll
  for (int i = 0; i < 2; i++)
#pragma unroll
    for (int j = 0; j < 4; j++) acc[i][j] = vzero;

  stage(0, 0);
  wait_async0();
  int par = 0;

#pragma unroll 1
  for (int k0 = 0; k0 < C_DIM; k0 += BK) {
    __syncthreads();                       // cur buf visible; other buf free to overwrite
    const int more = (k0 + BK < C_DIM);
    if (more) stage(par ^ 1, k0 + BK);     // async copies fly while WMMAs run

    Frag ah[2], al[2];
#pragma unroll
    for (int i = 0; i < 2; i++) {
      int rA = ((wn * 2 + i) * 16 + l2) * LDA;
      ah[i].u[0] = *(const uint4*)&sAh[par][rA + hf * 8];
      ah[i].u[1] = *(const uint4*)&sAh[par][rA + 16 + hf * 8];
      if (SPLIT) {
        al[i].u[0] = *(const uint4*)&sAl[par][rA + hf * 8];
        al[i].u[1] = *(const uint4*)&sAl[par][rA + 16 + hf * 8];
      }
    }
#pragma unroll
    for (int j = 0; j < 4; j++) {
      Frag bh, bl;
      int rB = ((wm * 4 + j) * 16 + l2) * LDA;
      bh.u[0] = *(const uint4*)&sBh[par][rB + hf * 8];
      bh.u[1] = *(const uint4*)&sBh[par][rB + 16 + hf * 8];
      if (SPLIT) {
        bl.u[0] = *(const uint4*)&sBl[par][rB + hf * 8];
        bl.u[1] = *(const uint4*)&sBl[par][rB + 16 + hf * 8];
      }
#pragma unroll
      for (int i = 0; i < 2; i++) {
        acc[i][j] = __builtin_amdgcn_wmma_f32_16x16x32_bf16(
            false, ah[i].v, false, bh.v, (short)0, acc[i][j], false, false);
        if (SPLIT) {
          acc[i][j] = __builtin_amdgcn_wmma_f32_16x16x32_bf16(
              false, ah[i].v, false, bl.v, (short)0, acc[i][j], false, false);
          acc[i][j] = __builtin_amdgcn_wmma_f32_16x16x32_bf16(
              false, al[i].v, false, bh.v, (short)0, acc[i][j], false, false);
        }
      }
    }

    if (more) wait_async0();               // staged buffer complete before next barrier
    par ^= 1;
  }

  // epilogue: d2 = f2 + m2 - 2*dot, clamp, min over m, atomicMin per n
#pragma unroll
  for (int i = 0; i < 2; i++) {
    int ngBase = nBase + (wn * 2 + i) * 16 + 8 * hf;
    float f2v[8];
#pragma unroll
    for (int r = 0; r < 8; r++) f2v[r] = f2[ngBase + r];
    float rmin[8];
#pragma unroll
    for (int r = 0; r < 8; r++) rmin[r] = 3.402823466e38f;
#pragma unroll
    for (int j = 0; j < 4; j++) {
      int mg = mBase + (wm * 4 + j) * 16 + l2;
      float m2v = m2[mg];
#pragma unroll
      for (int r = 0; r < 8; r++) {
        float d = f2v[r] + m2v - 2.0f * acc[i][j][r];
        d = fmaxf(d, 0.0f);
        rmin[r] = fminf(rmin[r], d);
      }
    }
#pragma unroll
    for (int r = 0; r < 8; r++) {
      float v = rmin[r];
      v = fminf(v, __shfl_xor(v, 1, 32));
      v = fminf(v, __shfl_xor(v, 2, 32));
      v = fminf(v, __shfl_xor(v, 4, 32));
      v = fminf(v, __shfl_xor(v, 8, 32));   // offsets <=8 keep halves (distinct n) separate
      if (l2 == 0) atomicMin(minbits + ngBase + r, __float_as_uint(v));
    }
  }
}

// ---------------- tail ----------------
__global__ void k_score(const unsigned* __restrict__ minbits, float* __restrict__ score, int n) {
  int i = blockIdx.x * blockDim.x + threadIdx.x;
  if (i < n) score[i] = sqrtf(fmaxf(__uint_as_float(minbits[i]), 0.f));
}

__global__ void k_imgmax(const float* __restrict__ score, float* __restrict__ out) {
  int b = blockIdx.x;
  float m = -3.402823466e38f;
  for (int i = threadIdx.x; i < FH_ * FW_; i += blockDim.x)
    m = fmaxf(m, score[b * FH_ * FW_ + i]);
#pragma unroll
  for (int off = 16; off >= 1; off >>= 1) m = fmaxf(m, __shfl_xor(m, off, 32));
  __shared__ float ls[8];
  int w = threadIdx.x >> 5;
  if ((threadIdx.x & 31) == 0) ls[w] = m;
  __syncthreads();
  if (threadIdx.x == 0) {
    float r = ls[0];
#pragma unroll
    for (int i = 1; i < 8; i++) r = fmaxf(r, ls[i]);
    out[b] = r;
  }
}

__global__ void k_resize(const float* __restrict__ score, float* __restrict__ out) {
  int idx = blockIdx.x * blockDim.x + threadIdx.x;
  if (idx >= B_IMG * IMG_ * IMG_) return;
  int x = idx % IMG_, y = (idx / IMG_) % IMG_, b = idx / (IMG_ * IMG_);
  float sx = (x + 0.5f) * ((float)FW_ / IMG_) - 0.5f;
  float sy = (y + 0.5f) * ((float)FH_ / IMG_) - 0.5f;
  int x0 = (int)floorf(sx), y0 = (int)floorf(sy);
  float wx = sx - x0, wy = sy - y0;
  int x0c = min(max(x0, 0), FW_ - 1), x1c = min(max(x0 + 1, 0), FW_ - 1);
  int y0c = min(max(y0, 0), FH_ - 1), y1c = min(max(y0 + 1, 0), FH_ - 1);
  const float* p = score + b * FH_ * FW_;
  float v = (1.f - wy) * ((1.f - wx) * p[y0c * FW_ + x0c] + wx * p[y0c * FW_ + x1c])
          +        wy  * ((1.f - wx) * p[y1c * FW_ + x0c] + wx * p[y1c * FW_ + x1c]);
  out[idx] = v;
}

__device__ __forceinline__ int refl_idx(int i, int n) {
  if (i < 0) i = -i;
  if (i >= n) i = 2 * n - 2 - i;
  return i;
}

__global__ void k_blur(const float* __restrict__ in, float* __restrict__ out, int horiz) {
  int idx = blockIdx.x * blockDim.x + threadIdx.x;
  if (idx >= B_IMG * IMG_ * IMG_) return;
  int x = idx % IMG_, y = (idx / IMG_) % IMG_, b = idx / (IMG_ * IMG_);
  float g[KS_]; float s = 0.f;
#pragma unroll
  for (int i = 0; i < KS_; i++) {
    float d = (float)i - (KS_ - 1) * 0.5f;
    g[i] = expf(-(d * d) / (2.f * SIGMA_ * SIGMA_));
    s += g[i];
  }
  float inv = 1.f / s;
  const float* p = in + b * IMG_ * IMG_;
  float acc = 0.f;
  if (horiz) {
#pragma unroll
    for (int i = 0; i < KS_; i++) acc += g[i] * p[y * IMG_ + refl_idx(x + i - PAD_R, IMG_)];
  } else {
#pragma unroll
    for (int i = 0; i < KS_; i++) acc += g[i] * p[refl_idx(y + i - PAD_R, IMG_) * IMG_ + x];
  }
  out[idx] = acc * inv;
}

// ---------------- host ----------------
extern "C" void kernel_launch(void* const* d_in, const int* in_sizes, int n_in,
                              void* d_out, int out_size, void* d_ws, size_t ws_size,
                              hipStream_t stream) {
  const float* features = (const float*)d_in[0];
  const float* memory   = (const float*)d_in[1];
  float* out = (float*)d_out;
  char* ws = (char*)d_ws;

  const size_t memB  = (size_t)M_PAD * C_DIM * sizeof(__hip_bfloat16);
  const size_t featB = (size_t)N_FEAT * C_DIM * sizeof(__hip_bfloat16);
  const size_t imgB  = (size_t)B_IMG * IMG_ * IMG_ * sizeof(float);

  struct Layout {
    size_t mem_hi, mem_lo, feat_hi, feat_lo, m2, f2, minb, score, rs, tmp, total;
  };
  auto mk = [&](bool split) {
    Layout L{}; size_t off = 0;
    auto take = [&](size_t bytes) { off = (off + 255) & ~(size_t)255; size_t p = off; off += bytes; return p; };
    L.mem_hi  = take(memB);
    L.feat_hi = take(featB);
    L.mem_lo  = split ? take(memB)  : L.mem_hi;
    L.feat_lo = split ? take(featB) : L.feat_hi;
    L.m2    = take((size_t)M_PAD * 4);
    L.f2    = take((size_t)N_FEAT * 4);
    L.minb  = take((size_t)N_FEAT * 4);
    L.score = take((size_t)N_FEAT * 4);
    L.rs    = take(imgB);
    L.tmp   = take(imgB);
    L.total = off;
    return L;
  };
  Layout Ls = mk(true);
  int split = (Ls.total <= ws_size) ? 1 : 0;
  Layout L = split ? Ls : mk(false);

  __hip_bfloat16* mem_hi  = (__hip_bfloat16*)(ws + L.mem_hi);
  __hip_bfloat16* mem_lo  = (__hip_bfloat16*)(ws + L.mem_lo);
  __hip_bfloat16* feat_hi = (__hip_bfloat16*)(ws + L.feat_hi);
  __hip_bfloat16* feat_lo = (__hip_bfloat16*)(ws + L.feat_lo);
  float*    m2    = (float*)(ws + L.m2);
  float*    f2    = (float*)(ws + L.f2);
  unsigned* minb  = (unsigned*)(ws + L.minb);
  float*    score = (float*)(ws + L.score);
  float*    rs    = (float*)(ws + L.rs);
  float*    tmp   = (float*)(ws + L.tmp);

  // 1) convert to bf16 hi/lo (pad rows -> zero)
  {
    long pad4 = (long)M_PAD * C_DIM / 4;
    k_convert<<<(unsigned)((pad4 + 255) / 256), 256, 0, stream>>>(
        memory, mem_hi, split ? mem_lo : nullptr, (long)M_MEM * C_DIM, (long)M_PAD * C_DIM);
    long f4 = (long)N_FEAT * C_DIM / 4;
    k_convert<<<(unsigned)((f4 + 255) / 256), 256, 0, stream>>>(
        features, feat_hi, split ? feat_lo : nullptr, (long)N_FEAT * C_DIM, (long)N_FEAT * C_DIM);
  }
  // 2) row norms (pad rows get 1e30 so they never win the min)
  k_rownorm<<<M_PAD, 128, 0, stream>>>(memory, m2, M_MEM, 1.0e30f);
  k_rownorm<<<N_FEAT, 128, 0, stream>>>(features, f2, N_FEAT, 0.f);
  // 3) init running min
  k_init_min<<<(N_FEAT + 255) / 256, 256, 0, stream>>>(minb, N_FEAT);
  // 4) fused GEMM + min  (n-blocks fastest so a memory slice is reused across all n while in L2)
  {
    dim3 grid(N_FEAT / BN, M_PAD / BM);   // 49 x 391
    if (split)
      k_gemm_min<1><<<grid, 256, 0, stream>>>(feat_hi, feat_lo, mem_hi, mem_lo, f2, m2, minb);
    else
      k_gemm_min<0><<<grid, 256, 0, stream>>>(feat_hi, feat_lo, mem_hi, mem_lo, f2, m2, minb);
  }
  // 5) tail
  k_score<<<(N_FEAT + 255) / 256, 256, 0, stream>>>(minb, score, N_FEAT);
  k_imgmax<<<B_IMG, 256, 0, stream>>>(score, out);                 // out[0..7]
  {
    int tot = B_IMG * IMG_ * IMG_;
    int blocks = (tot + 255) / 256;
    k_resize<<<blocks, 256, 0, stream>>>(score, rs);
    k_blur<<<blocks, 256, 0, stream>>>(rs, tmp, 1);
    k_blur<<<blocks, 256, 0, stream>>>(tmp, out + B_IMG, 0);       // masks
  }
}
